// MolGNN_90683939488465
// MI455X (gfx1250) — matrix operand order repaired and checked by reading, not compiled, not run on hardware
//
#include <hip/hip_runtime.h>
#include <hip/hip_bf16.h>

typedef __attribute__((ext_vector_type(16))) _Float16 v16h;
typedef __attribute__((ext_vector_type(8)))  float    v8f;

#define H96 96
#define FIN 32

__device__ inline v8f wmma32(v16h a, v16h b, v8f c) {
  // D = A(16x32 f16) * B(32x16 f16) + C(16x16 f32)
  return __builtin_amdgcn_wmma_f32_16x16x32_f16(
      /*neg_a=*/false, a, /*neg_b=*/false, b,
      /*c_mod=*/(short)0, c, /*reuse_a=*/false, /*reuse_b=*/false);
}

__device__ inline void atomic_add_f32(float* p, float v) {
  unsafeAtomicAdd(p, v);   // native global_atomic_add_f32 on gfx1250
}

__device__ inline float sigm(float x) { return 1.0f / (1.0f + __expf(-x)); }

// Load one A-fragment (16 halves/lane) for the 16x32 f16 A layout from a
// row-major f32 matrix: two contiguous 8-float runs per lane -> 4x b128 loads.
//   lanes 0-15: K = 0..7 and 16..23 ; lanes 16-31: K = 8..15 and 24..31
__device__ inline v16h load_a_frag(const float* __restrict__ A, size_t rowOff,
                                   int k0, int half) {
  const float4* p0 = (const float4*)(A + rowOff + k0 + half * 8);
  const float4* p1 = (const float4*)(A + rowOff + k0 + 16 + half * 8);
  float4 u0 = p0[0], u1 = p0[1], u2 = p1[0], u3 = p1[1];
  v16h a;
  a[0]=(_Float16)u0.x; a[1]=(_Float16)u0.y; a[2]=(_Float16)u0.z; a[3]=(_Float16)u0.w;
  a[4]=(_Float16)u1.x; a[5]=(_Float16)u1.y; a[6]=(_Float16)u1.z; a[7]=(_Float16)u1.w;
  a[8]=(_Float16)u2.x; a[9]=(_Float16)u2.y; a[10]=(_Float16)u2.z; a[11]=(_Float16)u2.w;
  a[12]=(_Float16)u3.x; a[13]=(_Float16)u3.y; a[14]=(_Float16)u3.z; a[15]=(_Float16)u3.w;
  return a;
}

// ---------------------------------------------------------------------------
// Pre-pack an f32 weight matrix into exact WMMA B-fragment order:
//   out[(((nt*(K/32)) + kb)*32 + lane)*16 + e] = W(k, col)
//   col = nt*16 + lane%16 ; k = kb*32 + (lane/16)*8 + (e%8) + (e/8)*16
//   W(k,j) read at W[k*sk + j*sj]
// Each wave in the GEMM then loads its v16h as one contiguous 32-byte chunk.
// ---------------------------------------------------------------------------
__global__ void pack_b_frag(const float* __restrict__ W, _Float16* __restrict__ out,
                            int K, int Ncol, int sk, int sj)
{
  int t = blockIdx.x * blockDim.x + threadIdx.x;
  if (t >= K * Ncol) return;
  int e = t & 15;
  int r = t >> 4;
  int lane = r & 31; r >>= 5;
  int kblocks = K / 32;
  int kb = r % kblocks;
  int nt = r / kblocks;
  int col = nt * 16 + (lane & 15);
  int k = kb * 32 + (lane >> 4) * 8 + (e & 7) + (e >> 3) * 16;
  out[t] = (_Float16)W[(size_t)k * sk + (size_t)col * sj];
}

__global__ void init_h(const float* __restrict__ x, float* __restrict__ h, int Nn) {
  int i = blockIdx.x * blockDim.x + threadIdx.x;
  if (i < Nn * H96) {
    int n = i / H96, c = i % H96;
    h[i] = (c < FIN) ? x[(size_t)n * FIN + c] : 0.0f;
  }
}

// ---------------------------------------------------------------------------
// WMMA GEMM: C[M x Ncol] = act(A[M x K] @ B + bias), M % 16 == 0.
// Bfrag pre-packed in fragment order. One wave -> one 16x16 C tile.
// ---------------------------------------------------------------------------
template <bool BIAS, bool RELU>
__global__ __launch_bounds__(128)
void gemm_wmma(const float* __restrict__ A, const _Float16* __restrict__ Bfrag,
               const float* __restrict__ bias, float* __restrict__ C,
               int M, int K, int Ncol)
{
  const int lane = threadIdx.x;
  const int half = lane >> 4, c16 = lane & 15;
  const int mt = blockIdx.x * blockDim.y + threadIdx.y;
  const int nt = blockIdx.y;
  if (mt * 16 >= M) return;                 // uniform per wave
  const size_t rowOff = (size_t)(mt * 16 + c16) * K;
  const int col = nt * 16 + c16;
  const int kblocks = K / 32;
  const v16h* __restrict__ Bf = (const v16h*)Bfrag + (size_t)nt * kblocks * 32 + lane;

  v8f acc = {};
  for (int kb = 0; kb < kblocks; ++kb) {
    v16h a = load_a_frag(A, rowOff, kb * 32, half);
    v16h b = Bf[(size_t)kb * 32];
    acc = wmma32(a, b, acc);
  }
  float bv = 0.0f;
  if (BIAS) bv = bias[col];
#pragma unroll
  for (int v = 0; v < 8; ++v) {
    int r = mt * 16 + half * 8 + v;
    float val = acc[v];
    if (BIAS) val += bv;
    if (RELU) val = fmaxf(val, 0.0f);
    C[(size_t)r * Ncol + col] = val;
  }
}

// ---------------------------------------------------------------------------
// Edge scatter: agg[dst] += m[src]   (one wave per edge, 3 channels/lane)
// ---------------------------------------------------------------------------
__global__ void scatter_edges(const float* __restrict__ m, float* __restrict__ agg,
                              const int* __restrict__ src, const int* __restrict__ dst, int E)
{
  int t = blockIdx.x * blockDim.x + threadIdx.x;
  int e = t >> 5, lane = t & 31;
  if (e >= E) return;
  int s = src[e], d = dst[e];
  const float* mr = m + (size_t)s * H96;
  float* ar = agg + (size_t)d * H96;
#pragma unroll
  for (int j = 0; j < 3; ++j) {
    int c = lane + 32 * j;
    atomic_add_f32(ar + c, mr[c]);
  }
}

// ---------------------------------------------------------------------------
// Fused GRU cell: gi = agg@Wih^T + bih, gh = h@Whh^T + bhh, gate update.
// One wave owns a 16x16 tile of h; 6 WMMA accumulator tiles in VGPRs.
// wih/whh passed in packed fragment order (Ncol=288 -> 18 n-tiles, K=96).
// ---------------------------------------------------------------------------
__global__ __launch_bounds__(128)
void gru_wmma(const float* __restrict__ hcur, const float* __restrict__ agg,
              const _Float16* __restrict__ wihF, const _Float16* __restrict__ whhF,
              const float* __restrict__ bih, const float* __restrict__ bhh,
              float* __restrict__ hnxt, int M)
{
  const int lane = threadIdx.x;
  const int half = lane >> 4, c16 = lane & 15;
  const int mt = blockIdx.x * blockDim.y + threadIdx.y;
  const int cb = blockIdx.y;                 // 0..5 column tile of H=96
  if (mt * 16 >= M) return;
  const size_t rowOff = (size_t)(mt * 16 + c16) * H96;
  const int kblocks = H96 / 32;              // 3

  v8f acc_i[3] = {};
  v8f acc_h[3] = {};
  for (int kb = 0; kb < kblocks; ++kb) {
    v16h aA = load_a_frag(agg,  rowOff, kb * 32, half);
    v16h aH = load_a_frag(hcur, rowOff, kb * 32, half);
#pragma unroll
    for (int s = 0; s < 3; ++s) {            // r / z / n slabs
      int nt = s * 6 + cb;                   // n-tile within Ncol=288
      size_t fidx = ((size_t)nt * kblocks + kb) * 32 + lane;
      v16h bI = ((const v16h*)wihF)[fidx];
      v16h bH = ((const v16h*)whhF)[fidx];
      acc_i[s] = wmma32(aA, bI, acc_i[s]);
      acc_h[s] = wmma32(aH, bH, acc_h[s]);
    }
  }

  const int col = cb * 16 + c16;
  const float bi_r = bih[col], bi_z = bih[H96 + col], bi_n = bih[2 * H96 + col];
  const float bh_r = bhh[col], bh_z = bhh[H96 + col], bh_n = bhh[2 * H96 + col];
#pragma unroll
  for (int v = 0; v < 8; ++v) {
    int r = mt * 16 + half * 8 + v;
    float ir = acc_i[0][v] + bi_r, iz = acc_i[1][v] + bi_z, inn = acc_i[2][v] + bi_n;
    float hr = acc_h[0][v] + bh_r, hz = acc_h[1][v] + bh_z, hn  = acc_h[2][v] + bh_n;
    float rg = sigm(ir + hr);
    float zg = sigm(iz + hz);
    float ng = tanhf(inn + rg * hn);
    float ho = hcur[(size_t)r * H96 + col];
    hnxt[(size_t)r * H96 + col] = (1.0f - zg) * ng + zg * ho;
  }
}

// ---------------------------------------------------------------------------
// relu + segment-sum pooling (one wave per node)
// ---------------------------------------------------------------------------
__global__ void pool_nodes(const float* __restrict__ h, const int* __restrict__ batch,
                           float* __restrict__ pooled, float* __restrict__ counts, int Nn)
{
  int t = blockIdx.x * blockDim.x + threadIdx.x;
  int n = t >> 5, lane = t & 31;
  if (n >= Nn) return;
  int b = batch[n];
#pragma unroll
  for (int j = 0; j < 3; ++j) {
    int c = lane + 32 * j;
    float v = fmaxf(h[(size_t)n * H96 + c], 0.0f);
    atomic_add_f32(pooled + (size_t)b * H96 + c, v);
  }
  if (lane == 0) atomic_add_f32(counts + b, 1.0f);
}

__global__ void bn_mean(const float* __restrict__ pooled, const float* __restrict__ counts,
                        const float* __restrict__ gamma, const float* __restrict__ beta,
                        float* __restrict__ outp, int Gn)
{
  int i = blockIdx.x * blockDim.x + threadIdx.x;
  if (i < Gn * H96) {
    int g = i / H96, c = i % H96;
    float mean = pooled[i] / fmaxf(counts[g], 1.0f);
    outp[i] = mean * (gamma[c] * rsqrtf(1.0f + 1e-5f)) + beta[c];
  }
}

// final fc3: [G x 96] @ [96] + b -> [G]   (one wave per graph)
__global__ void fc3_reduce(const float* __restrict__ h2, const float* __restrict__ w,
                           const float* __restrict__ b, float* __restrict__ out, int Gn)
{
  int t = blockIdx.x * blockDim.x + threadIdx.x;
  int g = t >> 5, lane = t & 31;
  if (g >= Gn) return;
  float s = 0.0f;
#pragma unroll
  for (int j = 0; j < 3; ++j) s += h2[(size_t)g * H96 + lane + 32 * j] * w[lane + 32 * j];
#pragma unroll
  for (int off = 16; off > 0; off >>= 1) s += __shfl_down(s, off, 32);
  if (lane == 0) out[g] = s + b[0];
}

// ---------------------------------------------------------------------------
static inline size_t alignUp(size_t x) { return (x + 255) & ~(size_t)255; }

extern "C" void kernel_launch(void* const* d_in, const int* in_sizes, int n_in,
                              void* d_out, int out_size, void* d_ws, size_t ws_size,
                              hipStream_t stream)
{
  const float* x      = (const float*)d_in[0];
  const int*   ei     = (const int*)  d_in[1];
  const int*   batch  = (const int*)  d_in[2];
  const float* conv_w = (const float*)d_in[3];   // [6,96,96]
  const float* w_ih   = (const float*)d_in[4];   // [288,96]
  const float* w_hh   = (const float*)d_in[5];   // [288,96]
  const float* b_ih   = (const float*)d_in[6];
  const float* b_hh   = (const float*)d_in[7];
  const float* gamma  = (const float*)d_in[8];
  const float* beta   = (const float*)d_in[9];
  const float* fc1w   = (const float*)d_in[10];  // [384,96]
  const float* fc1b   = (const float*)d_in[11];
  const float* fc2w   = (const float*)d_in[12];  // [96,384]
  const float* fc2b   = (const float*)d_in[13];
  const float* fc3w   = (const float*)d_in[14];  // [96]
  const float* fc3b   = (const float*)d_in[15];
  float* out = (float*)d_out;

  const int Nn = in_sizes[0] / FIN;      // 100000
  const int E  = in_sizes[1] / 2;        // 400000
  const int Gn = out_size;               // 4096
  const int L  = 6;

  // ---- workspace carve-up ----
  char* ws = (char*)d_ws;
  size_t off = 0;
  const size_t nodeMat = alignUp((size_t)Nn * H96 * sizeof(float));
  float* h0   = (float*)(ws + off); off += nodeMat;
  float* h1   = (float*)(ws + off); off += nodeMat;
  float* mbuf = (float*)(ws + off); off += nodeMat;
  float* agg  = (float*)(ws + off); off += nodeMat;
  _Float16* convwF = (_Float16*)(ws + off); off += alignUp((size_t)L * H96 * H96 * 2);
  _Float16* wihF   = (_Float16*)(ws + off); off += alignUp((size_t)3 * H96 * H96 * 2);
  _Float16* whhF   = (_Float16*)(ws + off); off += alignUp((size_t)3 * H96 * H96 * 2);
  _Float16* fc1wF  = (_Float16*)(ws + off); off += alignUp((size_t)384 * H96 * 2);
  _Float16* fc2wF  = (_Float16*)(ws + off); off += alignUp((size_t)H96 * 384 * 2);
  float* pooled = (float*)(ws + off); off += alignUp((size_t)Gn * H96 * sizeof(float));
  float* counts = (float*)(ws + off); off += alignUp((size_t)Gn * sizeof(float));
  float* hbn    = (float*)(ws + off); off += alignUp((size_t)Gn * H96 * sizeof(float));
  float* h1fc   = (float*)(ws + off); off += alignUp((size_t)Gn * 384 * sizeof(float));
  float* h2fc   = (float*)(ws + off); off += alignUp((size_t)Gn * H96 * sizeof(float));
  (void)ws_size; (void)n_in;

  // ---- pack all weights into WMMA fragment order (tiny, once per launch) ----
  auto pack = [&](const float* W, _Float16* d, int K, int Ncol, int sk, int sj) {
    int n = K * Ncol;
    pack_b_frag<<<(n + 255) / 256, 256, 0, stream>>>(W, d, K, Ncol, sk, sj);
  };
  for (int i = 0; i < L; ++i)                         // B(k,j) = conv_w[i][k*96 + j]
    pack(conv_w + (size_t)i * H96 * H96, convwF + (size_t)i * H96 * H96, H96, H96, H96, 1);
  pack(w_ih, wihF, H96, 3 * H96, 1, H96);             // B = w_ih^T
  pack(w_hh, whhF, H96, 3 * H96, 1, H96);             // B = w_hh^T
  pack(fc1w, fc1wF, H96, 384, 1, H96);                // B = fc1_w^T
  pack(fc2w, fc2wF, 384, H96, 1, 384);                // B = fc2_w^T

  // ---- h = pad(x) ----
  init_h<<<((size_t)Nn * H96 + 255) / 256, 256, 0, stream>>>(x, h0, Nn);

  const int* src = ei;
  const int* dst = ei + E;
  const int Mtiles = (Nn + 15) / 16;
  dim3 gblk(32, 4);
  dim3 ggrid((Mtiles + 3) / 4, H96 / 16);

  // ---- 6 GatedGraphConv layers ----
  for (int i = 0; i < L; ++i) {
    float* hcur = (i & 1) ? h1 : h0;
    float* hnxt = (i & 1) ? h0 : h1;
    gemm_wmma<false, false><<<ggrid, gblk, 0, stream>>>(
        hcur, convwF + (size_t)i * H96 * H96, nullptr, mbuf, Nn, H96, H96);
    hipMemsetAsync(agg, 0, (size_t)Nn * H96 * sizeof(float), stream);
    scatter_edges<<<((size_t)E * 32 + 255) / 256, 256, 0, stream>>>(mbuf, agg, src, dst, E);
    gru_wmma<<<ggrid, gblk, 0, stream>>>(hcur, agg, wihF, whhF, b_ih, b_hh, hnxt, Nn);
  }
  float* hfin = (L & 1) ? h1 : h0;   // L=6 -> h0

  // ---- relu + mean pool + BN ----
  hipMemsetAsync(pooled, 0, (size_t)Gn * H96 * sizeof(float), stream);
  hipMemsetAsync(counts, 0, (size_t)Gn * sizeof(float), stream);
  pool_nodes<<<((size_t)Nn * 32 + 255) / 256, 256, 0, stream>>>(hfin, batch, pooled, counts, Nn);
  bn_mean<<<((size_t)Gn * H96 + 255) / 256, 256, 0, stream>>>(pooled, counts, gamma, beta, hbn, Gn);

  // ---- MLP head on WMMA ----
  const int Gtiles = (Gn + 15) / 16;
  dim3 fgrid1((Gtiles + 3) / 4, 384 / 16);
  gemm_wmma<true, true><<<fgrid1, gblk, 0, stream>>>(hbn, fc1wF, fc1b, h1fc, Gn, H96, 384);
  dim3 fgrid2((Gtiles + 3) / 4, H96 / 16);
  gemm_wmma<true, true><<<fgrid2, gblk, 0, stream>>>(h1fc, fc2wF, fc2b, h2fc, Gn, 384, H96);
  fc3_reduce<<<((size_t)Gn * 32 + 255) / 256, 256, 0, stream>>>(h2fc, fc3w, fc3b, out, Gn);
}
